// GNN_20796231647842
// MI455X (gfx1250) — matrix operand order repaired and checked by reading, not compiled, not run on hardware
//
#include <hip/hip_runtime.h>

#define N_NODES 50000
#define N_EDGES 800000
#define F_IN 32
#define H 128
#define N_REL 4
#define MAX_DEG 10
#define N_BUCKETS 11
#define N_GRAPHS 64

typedef __attribute__((ext_vector_type(16))) __bf16 v16bf;
typedef __attribute__((ext_vector_type(8)))  float  v8f;

// ---------------------------------------------------------------- helpers ---

__device__ __forceinline__ unsigned short f2bf(float f) {
    union { float f; unsigned u; } x; x.f = f;
    unsigned u = x.u + 0x7FFFu + ((x.u >> 16) & 1u);   // round-to-nearest-even
    return (unsigned short)(u >> 16);
}
__device__ __forceinline__ unsigned pk2(float a, float b) {
    return (unsigned)f2bf(a) | ((unsigned)f2bf(b) << 16);
}

union FragBF { v16bf v; uint4 q[2]; unsigned w[8]; };

// A-matrix 16x32 bf16 fragment from a bf16 row (ISA 7.12.2: lane L holds
// M=L%16; K split kb..kb+7 and 16+kb..16+kb+7, kb = (lane>=16)?8:0)
__device__ __forceinline__ v16bf load_fragA_bf16(const unsigned short* rowPtr, int k0, int lane) {
    int kb = (lane >> 4) << 3;
    FragBF f;
    f.q[0] = *(const uint4*)(rowPtr + k0 + kb);
    f.q[1] = *(const uint4*)(rowPtr + k0 + 16 + kb);
    return f.v;
}

// B-matrix 32x16 bf16 fragment from pre-transposed bf16 weight row [N][K]
// (lane = column N; K split kb..kb+7 and 16+kb..16+kb+7, kb = (lane>=16)?8:0)
__device__ __forceinline__ v16bf load_fragB_bf16(const unsigned short* rowPtr, int k0, int lane) {
    int koff = (lane >> 4) << 3;
    FragBF f;
    f.q[0] = *(const uint4*)(rowPtr + k0 + koff);
    f.q[1] = *(const uint4*)(rowPtr + k0 + koff + 16);
    return f.v;
}

__device__ __forceinline__ v8f wmma_bf16(v16bf a, v16bf b, v8f c) {
    return __builtin_amdgcn_wmma_f32_16x16x32_bf16(false, a, false, b, (short)0, c, false, false);
}

__device__ __forceinline__ v8f zero8() {
    v8f z;
    for (int i = 0; i < 8; ++i) z[i] = 0.f;
    return z;
}

// ---------------------------------------------------------------- kernels ---

__global__ void k_zero(unsigned* p, long n) {
    long i = (long)blockIdx.x * blockDim.x + threadIdx.x;
    long stride = (long)gridDim.x * blockDim.x;
    for (; i < n; i += stride) p[i] = 0u;
}

// flat f32 -> bf16 conversion (float4 -> uint2), optional fused relu
__global__ void k_cvt_bf16(const float* __restrict__ in, unsigned short* __restrict__ out,
                           long n4, int relu) {
    long i = (long)blockIdx.x * blockDim.x + threadIdx.x;
    long stride = (long)gridDim.x * blockDim.x;
    for (; i < n4; i += stride) {
        float4 v = ((const float4*)in)[i];
        if (relu) {
            v.x = fmaxf(v.x, 0.f); v.y = fmaxf(v.y, 0.f);
            v.z = fmaxf(v.z, 0.f); v.w = fmaxf(v.w, 0.f);
        }
        uint2 p;
        p.x = pk2(v.x, v.y);
        p.y = pk2(v.z, v.w);
        ((uint2*)out)[i] = p;
    }
}

// per-(node,rel) row: bf16 = f32 * 1/max(cnt,1)   (fuses RGCN mean)
__global__ void k_cvt_agg(const float* __restrict__ agg, const float* __restrict__ cnt,
                          unsigned short* __restrict__ out) {
    int lane = threadIdx.x & 31;
    long row = (long)blockIdx.x * 8 + (threadIdx.x >> 5);
    if (row >= (long)N_NODES * N_REL) return;
    float scale = 1.0f / fmaxf(cnt[row], 1.0f);
    float4 v = ((const float4*)(agg + row * H))[lane];
    uint2 p;
    p.x = pk2(v.x * scale, v.y * scale);
    p.y = pk2(v.z * scale, v.w * scale);
    ((uint2*)(out + row * H))[lane] = p;
}

// transpose+convert nmat matrices [K][Ncols] f32 -> [Ncols][K] bf16
__global__ void k_wconv(const float* __restrict__ src, unsigned short* __restrict__ dst,
                        int K, int Ncols, long total) {
    long gid = (long)blockIdx.x * blockDim.x + threadIdx.x;
    if (gid >= total) return;
    long per = (long)K * Ncols;
    long m = gid / per;
    long rem = gid - m * per;
    int k = (int)(rem / Ncols);
    int n = (int)(rem % Ncols);
    dst[m * per + (long)n * K + k] = f2bf(src[gid]);
}

__global__ void k_count(const int* __restrict__ dst, const int* __restrict__ rel,
                        float* __restrict__ cntR, int* __restrict__ degI) {
    int e = blockIdx.x * 256 + threadIdx.x;
    if (e >= N_EDGES) return;
    int d = dst[e];
    atomicAdd(&cntR[d * N_REL + rel[e]], 1.0f);
    atomicAdd(&degI[d], 1);
}

__global__ void k_bucket(const int* __restrict__ degI, int* __restrict__ bCnt,
                         int* __restrict__ bLst) {
    int n = blockIdx.x * 256 + threadIdx.x;
    if (n >= N_NODES) return;
    int d = degI[n]; d = d > MAX_DEG ? MAX_DEG : d;
    int pos = atomicAdd(&bCnt[d], 1);
    bLst[d * N_NODES + pos] = n;
}

// per-edge feature scatter with relation: agg[(dst*4+rel)*H] += relu?(h[src*H])
__global__ void k_scatter_rel(const float* __restrict__ h, const int* __restrict__ src,
                              const int* __restrict__ dst, const int* __restrict__ rel,
                              float* __restrict__ agg, int relu) {
    int lane = threadIdx.x & 31;
    int e = blockIdx.x * 8 + (threadIdx.x >> 5);
    if (e >= N_EDGES) return;
    int s = src[e], d = dst[e], rl = rel[e];
    float4 v = ((const float4*)(h + (size_t)s * H))[lane];
    if (relu) {
        v.x = fmaxf(v.x, 0.f); v.y = fmaxf(v.y, 0.f);
        v.z = fmaxf(v.z, 0.f); v.w = fmaxf(v.w, 0.f);
    }
    float* ap = agg + ((size_t)d * N_REL + rl) * H + lane * 4;
    atomicAdd(ap + 0, v.x); atomicAdd(ap + 1, v.y);
    atomicAdd(ap + 2, v.z); atomicAdd(ap + 3, v.w);
}

__global__ void k_scatter_plain(const float* __restrict__ h, const int* __restrict__ src,
                                const int* __restrict__ dst, float* __restrict__ agg) {
    int lane = threadIdx.x & 31;
    int e = blockIdx.x * 8 + (threadIdx.x >> 5);
    if (e >= N_EDGES) return;
    int s = src[e], d = dst[e];
    float4 v = ((const float4*)(h + (size_t)s * H))[lane];
    float* ap = agg + (size_t)d * H + lane * 4;
    atomicAdd(ap + 0, v.x); atomicAdd(ap + 1, v.y);
    atomicAdd(ap + 2, v.z); atomicAdd(ap + 3, v.w);
}

// h = x[50000,32] @ W[32,128] + b  (pure bf16 operands; dual store f32 + bf16)
__global__ void __launch_bounds__(256)
k_embed(const unsigned short* __restrict__ xb, const unsigned short* __restrict__ Wt,
        const float* __restrict__ bias, float* __restrict__ out,
        unsigned short* __restrict__ outb) {
    int lane = threadIdx.x & 31, wv = threadIdx.x >> 5;
    int node0 = blockIdx.x * 16, n0 = wv * 16, r = lane & 15;
    v8f acc = zero8();
    v16bf a = load_fragA_bf16(xb + (size_t)(node0 + r) * F_IN, 0, lane);
    v16bf b = load_fragB_bf16(Wt + (size_t)(n0 + r) * F_IN, 0, lane);
    acc = wmma_bf16(a, b, acc);
    int col = n0 + r, hi = lane >> 4;
    float bv = bias[col];
    for (int i = 0; i < 8; ++i) {
        int m = i + 8 * hi;
        float v = acc[i] + bv;
        out [(size_t)(node0 + m) * H + col] = v;
        outb[(size_t)(node0 + m) * H + col] = f2bf(v);
    }
}

// out = relu( sum_r (agg_r/cnt_r) @ Wrel_r + h @ Wroot + b )   20 WMMAs/wave
__global__ void __launch_bounds__(256)
k_rgcn(const unsigned short* __restrict__ hb, const unsigned short* __restrict__ aggb,
       const unsigned short* __restrict__ WrelT, const unsigned short* __restrict__ WrootT,
       const float* __restrict__ bias, float* __restrict__ out,
       unsigned short* __restrict__ outb) {
    int lane = threadIdx.x & 31, wv = threadIdx.x >> 5;
    int node0 = blockIdx.x * 16, n0 = wv * 16, r = lane & 15;
    v8f acc = zero8();
#pragma unroll
    for (int rel = 0; rel < N_REL; ++rel) {
        const unsigned short* arow = aggb + ((size_t)(node0 + r) * N_REL + rel) * H;
        const unsigned short* wrow = WrelT + (size_t)rel * H * H + (size_t)(n0 + r) * H;
#pragma unroll
        for (int k0 = 0; k0 < H; k0 += 32) {
            v16bf a = load_fragA_bf16(arow, k0, lane);
            v16bf b = load_fragB_bf16(wrow, k0, lane);
            acc = wmma_bf16(a, b, acc);
        }
    }
    const unsigned short* hrow = hb + (size_t)(node0 + r) * H;
    const unsigned short* wroot = WrootT + (size_t)(n0 + r) * H;
#pragma unroll
    for (int k0 = 0; k0 < H; k0 += 32) {
        v16bf a = load_fragA_bf16(hrow, k0, lane);
        v16bf b = load_fragB_bf16(wroot, k0, lane);
        acc = wmma_bf16(a, b, acc);
    }
    int col = n0 + r, hi = lane >> 4;
    float bv = bias[col];
    for (int i = 0; i < 8; ++i) {
        int m = i + 8 * hi;
        float v = fmaxf(acc[i] + bv, 0.0f);           // fused relu
        out [(size_t)(node0 + m) * H + col] = v;      // f32 for scatter/pool
        outb[(size_t)(node0 + m) * H + col] = f2bf(v); // bf16 for MF root operand
    }
}

// degree-bucketed MFConv: out[n] = agg[n]@Wl[d] + h[n]@Wr[d] + bl[d]   8 WMMAs/wave
__global__ void __launch_bounds__(256)
k_mf(const unsigned short* __restrict__ aggb, const unsigned short* __restrict__ hb,
     const unsigned short* __restrict__ WlT, const unsigned short* __restrict__ WrT,
     const float* __restrict__ bl, const int* __restrict__ bCnt,
     const int* __restrict__ bLst, float* __restrict__ out) {
    int d = blockIdx.y;
    int count = bCnt[d];
    int start = blockIdx.x * 16;
    if (start >= count) return;
    int lane = threadIdx.x & 31, wv = threadIdx.x >> 5;
    int n0 = wv * 16, r = lane & 15;
    int rowIdx = start + r; if (rowIdx >= count) rowIdx = count - 1;
    int node = bLst[d * N_NODES + rowIdx];
    const unsigned short* wl = WlT + (size_t)d * H * H + (size_t)(n0 + r) * H;
    const unsigned short* wr = WrT + (size_t)d * H * H + (size_t)(n0 + r) * H;
    const unsigned short* arow = aggb + (size_t)node * H;
    const unsigned short* hrow = hb + (size_t)node * H;
    v8f acc = zero8();
#pragma unroll
    for (int k0 = 0; k0 < H; k0 += 32) {
        v16bf a0 = load_fragA_bf16(arow, k0, lane);
        v16bf b0 = load_fragB_bf16(wl, k0, lane);
        acc = wmma_bf16(a0, b0, acc);
        v16bf a1 = load_fragA_bf16(hrow, k0, lane);
        v16bf b1 = load_fragB_bf16(wr, k0, lane);
        acc = wmma_bf16(a1, b1, acc);
    }
    int col = n0 + r, hi = lane >> 4;
    float bv = bl[d * H + col];
    for (int i = 0; i < 8; ++i) {
        int m = i + 8 * hi;
        int rowI = start + m;
        if (rowI < count) {
            int nodeS = bLst[d * N_NODES + rowI];
            out[(size_t)nodeS * H + col] = acc[i] + bv;
        }
    }
}

__global__ void k_pool(const float* __restrict__ h, const int* __restrict__ batch,
                       float* __restrict__ g) {
    int lane = threadIdx.x & 31;
    int n = blockIdx.x * 8 + (threadIdx.x >> 5);
    if (n >= N_NODES) return;
    int b = batch[n];
    float4 v = ((const float4*)(h + (size_t)n * H))[lane];
    float* gp = g + (size_t)b * H + lane * 4;
    atomicAdd(gp + 0, v.x); atomicAdd(gp + 1, v.y);
    atomicAdd(gp + 2, v.z); atomicAdd(gp + 3, v.w);
}

// t = relu(g[64,128] @ lin1 + b1)   4 WMMAs/wave
__global__ void __launch_bounds__(256)
k_head1(const unsigned short* __restrict__ gb, const unsigned short* __restrict__ lin1T,
        const float* __restrict__ b1, float* __restrict__ t) {
    int lane = threadIdx.x & 31, wv = threadIdx.x >> 5;
    int m0 = blockIdx.x * 16, n0 = wv * 16, r = lane & 15;
    v8f acc = zero8();
#pragma unroll
    for (int k0 = 0; k0 < H; k0 += 32) {
        v16bf a = load_fragA_bf16(gb + (size_t)(m0 + r) * H, k0, lane);
        v16bf b = load_fragB_bf16(lin1T + (size_t)(n0 + r) * H, k0, lane);
        acc = wmma_bf16(a, b, acc);
    }
    int col = n0 + r, hi = lane >> 4;
    float bv = b1[col];
    for (int i = 0; i < 8; ++i) {
        int m = i + 8 * hi;
        t[(size_t)(m0 + m) * H + col] = fmaxf(acc[i] + bv, 0.0f);
    }
}

__global__ void k_head2(const float* __restrict__ t, const float* __restrict__ w2,
                        const float* __restrict__ b2, float* __restrict__ out) {
    int gI = threadIdx.x;
    if (gI >= N_GRAPHS) return;
    float s = b2[0];
    for (int f = 0; f < H; ++f) s += t[gI * H + f] * w2[f];
    out[gI] = s;
}

// ------------------------------------------------------------------ launch ---

extern "C" void kernel_launch(void* const* d_in, const int* in_sizes, int n_in,
                              void* d_out, int out_size, void* d_ws, size_t ws_size,
                              hipStream_t stream) {
    const float* x     = (const float*)d_in[0];
    const int*   eidx  = (const int*)d_in[1];
    const int*   eattr = (const int*)d_in[2];
    const int*   batch = (const int*)d_in[3];
    const float* embW  = (const float*)d_in[4];
    const float* embB  = (const float*)d_in[5];
    const float* Wrel  = (const float*)d_in[6];
    const float* Wroot = (const float*)d_in[7];
    const float* rgcnB = (const float*)d_in[8];
    const float* Wl    = (const float*)d_in[9];
    const float* bl    = (const float*)d_in[10];
    const float* Wr    = (const float*)d_in[11];
    const float* lin1W = (const float*)d_in[12];
    const float* lin1B = (const float*)d_in[13];
    const float* lin2W = (const float*)d_in[14];
    const float* lin2B = (const float*)d_in[15];
    const int* src = eidx;
    const int* dst = eidx + N_EDGES;

    char* ws = (char*)d_ws;
    size_t off = 0;
    auto grab = [&](size_t bytes) -> void* {
        void* p = ws + off;
        off += (bytes + 255) & ~(size_t)255;
        return p;
    };
    float* hA    = (float*)grab((size_t)N_NODES * H * 4);
    float* hB    = (float*)grab((size_t)N_NODES * H * 4);
    float* aggR  = (float*)grab((size_t)N_NODES * N_REL * H * 4);
    float* cntR  = (float*)grab((size_t)N_NODES * N_REL * 4);
    float* aggM  = (float*)grab((size_t)N_NODES * H * 4);
    int*   degI  = (int*)grab((size_t)N_NODES * 4);
    int*   bCnt  = (int*)grab(N_BUCKETS * 4);
    int*   bLst  = (int*)grab((size_t)N_BUCKETS * N_NODES * 4);
    float* gp    = (float*)grab((size_t)N_GRAPHS * H * 4);
    float* t1    = (float*)grab((size_t)N_GRAPHS * H * 4);
    unsigned short* xb     = (unsigned short*)grab((size_t)N_NODES * F_IN * 2);
    unsigned short* hAb    = (unsigned short*)grab((size_t)N_NODES * H * 2);
    unsigned short* hBb    = (unsigned short*)grab((size_t)N_NODES * H * 2);
    unsigned short* aggRb  = (unsigned short*)grab((size_t)N_NODES * N_REL * H * 2);
    unsigned short* aggMb  = (unsigned short*)grab((size_t)N_NODES * H * 2);
    unsigned short* gpb    = (unsigned short*)grab((size_t)N_GRAPHS * H * 2);
    unsigned short* embWt  = (unsigned short*)grab((size_t)H * F_IN * 2);
    unsigned short* WrelT  = (unsigned short*)grab((size_t)2 * N_REL * H * H * 2);
    unsigned short* WrootT = (unsigned short*)grab((size_t)2 * H * H * 2);
    unsigned short* WlT    = (unsigned short*)grab((size_t)2 * N_BUCKETS * H * H * 2);
    unsigned short* WrT    = (unsigned short*)grab((size_t)2 * N_BUCKETS * H * H * 2);
    unsigned short* lin1T  = (unsigned short*)grab((size_t)H * H * 2);

    // --- per-call structure: counts, degrees, buckets, bf16 weights/inputs
    k_zero<<<256, 256, 0, stream>>>((unsigned*)cntR, (long)N_NODES * N_REL);
    k_zero<<<256, 256, 0, stream>>>((unsigned*)degI, (long)N_NODES);
    k_zero<<<1, 32, 0, stream>>>((unsigned*)bCnt, (long)N_BUCKETS);
    k_count<<<(N_EDGES + 255) / 256, 256, 0, stream>>>(dst, eattr, cntR, degI);
    k_bucket<<<(N_NODES + 255) / 256, 256, 0, stream>>>(degI, bCnt, bLst);

    k_wconv<<<(F_IN * H + 255) / 256, 256, 0, stream>>>(embW, embWt, F_IN, H, (long)F_IN * H);
    k_wconv<<<(2 * N_REL * H * H + 255) / 256, 256, 0, stream>>>(Wrel, WrelT, H, H, (long)2 * N_REL * H * H);
    k_wconv<<<(2 * H * H + 255) / 256, 256, 0, stream>>>(Wroot, WrootT, H, H, (long)2 * H * H);
    k_wconv<<<(2 * N_BUCKETS * H * H + 255) / 256, 256, 0, stream>>>(Wl, WlT, H, H, (long)2 * N_BUCKETS * H * H);
    k_wconv<<<(2 * N_BUCKETS * H * H + 255) / 256, 256, 0, stream>>>(Wr, WrT, H, H, (long)2 * N_BUCKETS * H * H);
    k_wconv<<<(H * H + 255) / 256, 256, 0, stream>>>(lin1W, lin1T, H, H, (long)H * H);
    k_cvt_bf16<<<2048, 256, 0, stream>>>(x, xb, (long)N_NODES * F_IN / 4, 0);

    // --- embed (writes f32 hA + bf16 hAb)
    k_embed<<<N_NODES / 16, 256, 0, stream>>>(xb, embWt, embB, hA, hAb);

    // --- 2 GNN blocks
    for (int blk = 0; blk < 2; ++blk) {
        int ri = (blk > 0) ? 1 : 0;
        if (ri)  // refresh bf16 copy of hA with relu fused (hA itself stays raw)
            k_cvt_bf16<<<4096, 256, 0, stream>>>(hA, hAb, (long)N_NODES * H / 4, 1);
        k_zero<<<2048, 256, 0, stream>>>((unsigned*)aggR, (long)N_NODES * N_REL * H);
        k_scatter_rel<<<N_EDGES / 8, 256, 0, stream>>>(hA, src, dst, eattr, aggR, ri);
        k_cvt_agg<<<(N_NODES * N_REL + 7) / 8, 256, 0, stream>>>(aggR, cntR, aggRb);
        k_rgcn<<<N_NODES / 16, 256, 0, stream>>>(hAb, aggRb,
            WrelT + (size_t)blk * N_REL * H * H, WrootT + (size_t)blk * H * H,
            rgcnB + blk * H, hB, hBb);
        k_zero<<<2048, 256, 0, stream>>>((unsigned*)aggM, (long)N_NODES * H);
        k_scatter_plain<<<N_EDGES / 8, 256, 0, stream>>>(hB, src, dst, aggM);
        k_cvt_bf16<<<4096, 256, 0, stream>>>(aggM, aggMb, (long)N_NODES * H / 4, 0);
        k_mf<<<dim3(N_NODES / 16, N_BUCKETS), 256, 0, stream>>>(aggMb, hBb,
            WlT + (size_t)blk * N_BUCKETS * H * H, WrT + (size_t)blk * N_BUCKETS * H * H,
            bl + blk * N_BUCKETS * H, bCnt, bLst, hA);
    }

    // --- pool + head
    k_zero<<<32, 256, 0, stream>>>((unsigned*)gp, (long)N_GRAPHS * H);
    k_pool<<<(N_NODES + 7) / 8, 256, 0, stream>>>(hA, batch, gp);
    k_cvt_bf16<<<8, 256, 0, stream>>>(gp, gpb, (long)N_GRAPHS * H / 4, 0);
    k_head1<<<N_GRAPHS / 16, 256, 0, stream>>>(gpb, lin1T, lin1B, t1);
    k_head2<<<1, 64, 0, stream>>>(t1, lin2W, lin2B, (float*)d_out);

    (void)in_sizes; (void)n_in; (void)out_size; (void)ws_size;
}
